// VQNet_8873402434290
// MI455X (gfx1250) — matrix operand order repaired and checked by reading, not compiled, not run on hardware
//
#include <hip/hip_runtime.h>
#include <hip/hip_bf16.h>
#include <stdint.h>

typedef __bf16 bf16_t;
typedef __attribute__((ext_vector_type(16))) __bf16 v16bf;
typedef __attribute__((ext_vector_type(8)))  __bf16 v8bf;
typedef __attribute__((ext_vector_type(4)))  __bf16 v4bf;
typedef __attribute__((ext_vector_type(8)))  float  v8f;

#define N_TOKENS   32768
#define N_EMBED    8192
#define EMBED_DIM  512
#define KSTEPS     (EMBED_DIM / 32)
#define BLOCK_TOKENS 128            // tokens per workgroup (8 waves x 16)
#define CHUNK_CODES  32             // codes staged per iteration
#define XPAD 8                      // bf16 pad -> row stride 520 (bank-conflict free)
#define XSTRIDE (EMBED_DIM + XPAD)
#define EBUF (CHUNK_CODES * XSTRIDE)   // one E buffer, bf16 elems

// Async global->LDS 16-byte copy: no data VGPRs, tracked by ASYNCcnt.
__device__ __forceinline__ void async_copy_b128(unsigned lds_byte_off, const void* gptr) {
    asm volatile("global_load_async_to_lds_b128 %0, %1, off"
                 :
                 : "v"(lds_byte_off), "v"((unsigned long long)(uintptr_t)gptr)
                 : "memory");
}
__device__ __forceinline__ void wait_async0() {
    asm volatile("s_wait_asynccnt 0x0" ::: "memory");
}

// A-fragment pair load for one k-step (16-bit 16x32 A layout).
__device__ __forceinline__ void load_afrags(v16bf& a0, v16bf& a1,
                                            const bf16_t* ea0, const bf16_t* ea1,
                                            int kb, int hi) {
    ((v8bf*)&a0)[0] = *(const v8bf*)(ea0 + kb + hi * 8);
    ((v8bf*)&a0)[1] = *(const v8bf*)(ea0 + kb + 16 + hi * 8);
    ((v8bf*)&a1)[0] = *(const v8bf*)(ea1 + kb + hi * 8);
    ((v8bf*)&a1)[1] = *(const v8bf*)(ea1 + kb + 16 + hi * 8);
}

// ---------- kernel 1: norms + f32->bf16 conversion of the code table ----------
__global__ __launch_bounds__(256) void vq_prep(const float* __restrict__ E,
                                               bf16_t* __restrict__ E16,
                                               float* __restrict__ norms) {
    int wave = threadIdx.x >> 5;
    int lane = threadIdx.x & 31;
    int code = blockIdx.x * 8 + wave;
    const float* row  = E   + (size_t)code * EMBED_DIM;
    bf16_t*     row16 = E16 + (size_t)code * EMBED_DIM;
    float s = 0.f;
    for (int k = lane * 4; k < EMBED_DIM; k += 32 * 4) {
        float4 v = *(const float4*)(row + k);
        s += v.x * v.x + v.y * v.y + v.z * v.z + v.w * v.w;
        v4bf p = { (bf16_t)v.x, (bf16_t)v.y, (bf16_t)v.z, (bf16_t)v.w };
        *(v4bf*)(row16 + k) = p;
    }
    #pragma unroll
    for (int off = 16; off > 0; off >>= 1)
        s += __shfl_down(s, off, 32);
    if (lane == 0) norms[code] = s;
}

// ---------- kernel 2: distances (WMMA bf16) + argmin + gather ----------
__global__ __launch_bounds__(256, 1) void vq_main(const float* __restrict__ X,
                                                  const float* __restrict__ E,
                                                  const bf16_t* __restrict__ E16,
                                                  const float* __restrict__ norms,
                                                  float* __restrict__ Out) {
    extern __shared__ char smem_raw[];
    // Phase-1 overlay: X tile [128][520] bf16 (133,120 B)
    bf16_t* Xs = (bf16_t*)smem_raw;
    // Phase-2 overlay (same memory): E double buffer + winners
    bf16_t* Es   = (bf16_t*)smem_raw;                 // [2][32][520] bf16
    int*    IdxS = (int*)(Es + 2 * EBUF);             // [128]

    const int tid  = threadIdx.x;
    const int wave = tid >> 5;
    const int lane = tid & 31;
    const int tokBase = blockIdx.x * BLOCK_TOKENS;

    const int nloc  = lane & 15;      // token column within wave tile
    const int hi    = lane >> 4;      // half-wave selector
    const int myTok = wave * 16;

    // ---- phase 1: stage X tile f32->bf16 into LDS, then lift B-fragments to VGPRs ----
    for (int i = tid; i < BLOCK_TOKENS * (EMBED_DIM / 4); i += 256) {
        int r = i / (EMBED_DIM / 4);
        int c = (i % (EMBED_DIM / 4)) * 4;
        float4 v = *(const float4*)(X + (size_t)(tokBase + r) * EMBED_DIM + c);
        v4bf p = { (bf16_t)v.x, (bf16_t)v.y, (bf16_t)v.z, (bf16_t)v.w };
        *(v4bf*)(Xs + r * XSTRIDE + c) = p;
    }
    __syncthreads();

    // Persistent B fragments: 16 k-steps x v16bf = 128 VGPRs, live whole kernel.
    // B 32x16 bf16 layout: N = lane%16, K = hi*8 + [0..7] and 16 + hi*8 + [0..7].
    v16bf bfrag[KSTEPS];
    {
        const bf16_t* xrow = Xs + (myTok + nloc) * XSTRIDE + hi * 8;
        #pragma unroll
        for (int ks = 0; ks < KSTEPS; ++ks) {
            ((v8bf*)&bfrag[ks])[0] = *(const v8bf*)(xrow + ks * 32);
            ((v8bf*)&bfrag[ks])[1] = *(const v8bf*)(xrow + ks * 32 + 16);
        }
    }
    __syncthreads();   // all B-frags read before overlay is overwritten

    // ---- stage chunk 0 into buffer 0 via async global->LDS (no data VGPRs) ----
    {
        const unsigned dstBase = (unsigned)(uintptr_t)Es;
        const char* srcBase = (const char*)E16;
        #pragma unroll
        for (int t = 0; t < 8; ++t) {
            int i = tid + t * 256;              // 2048 x 16B = 32 codes x 512 bf16
            int r = i >> 6, c = i & 63;
            async_copy_b128(dstBase + (unsigned)(r * XSTRIDE + c * 8) * 2u,
                            srcBase + (size_t)i * 16);
        }
    }
    wait_async0();
    __syncthreads();

    float minv = 3.402823466e+38f;
    int   mini = 0x7fffffff;
    int   buf  = 0;

    for (int cc = 0; cc < N_EMBED; cc += CHUNK_CODES) {
        // -- issue next chunk's async copies into the other buffer (overlaps WMMA) --
        const bool haveNext = (cc + CHUNK_CODES) < N_EMBED;
        if (haveNext) {
            const unsigned dstBase = (unsigned)(uintptr_t)(Es + (buf ^ 1) * EBUF);
            const char* srcBase = (const char*)(E16 + (size_t)(cc + CHUNK_CODES) * EMBED_DIM);
            #pragma unroll
            for (int t = 0; t < 8; ++t) {
                int i = tid + t * 256;
                int r = i >> 6, c = i & 63;
                async_copy_b128(dstBase + (unsigned)(r * XSTRIDE + c * 8) * 2u,
                                srcBase + (size_t)i * 16);
            }
        }

        // -- per-half-wave norm vectors: contiguous, broadcast loads --
        float4 n0a = *(const float4*)(norms + cc + hi * 8);
        float4 n0b = *(const float4*)(norms + cc + hi * 8 + 4);
        float4 n1a = *(const float4*)(norms + cc + 16 + hi * 8);
        float4 n1b = *(const float4*)(norms + cc + 16 + hi * 8 + 4);

        // -- compute: 2 M-tiles x 16 k-steps, A-fragments pipelined 1 step ahead --
        const bf16_t* Ecur = Es + buf * EBUF;
        const bf16_t* ea0 = Ecur + nloc * XSTRIDE;          // codes 0..15
        const bf16_t* ea1 = Ecur + (nloc + 16) * XSTRIDE;   // codes 16..31
        v8f acc0 = {};
        v8f acc1 = {};
        v16bf a0c, a1c, a0n, a1n;
        load_afrags(a0c, a1c, ea0, ea1, 0, hi);
        #pragma unroll
        for (int ks = 0; ks < KSTEPS; ++ks) {
            if (ks + 1 < KSTEPS)
                load_afrags(a0n, a1n, ea0, ea1, (ks + 1) * 32, hi);
            acc0 = __builtin_amdgcn_wmma_f32_16x16x32_bf16(
                false, a0c, false, bfrag[ks], (short)0, acc0, false, false);
            acc1 = __builtin_amdgcn_wmma_f32_16x16x32_bf16(
                false, a1c, false, bfrag[ks], (short)0, acc1, false, false);
            a0c = a0n;   // register rename after unroll
            a1c = a1n;
        }

        // -- dist = ||e||^2 - 2*x.e ; running argmin (first-index tie-break) --
        #pragma unroll
        for (int r = 0; r < 8; ++r) {
            float nv0 = (r < 4) ? (&n0a.x)[r] : (&n0b.x)[r - 4];
            float nv1 = (r < 4) ? (&n1a.x)[r] : (&n1b.x)[r - 4];
            int mrow = hi * 8 + r;
            float d0 = nv0 - 2.0f * acc0[r];
            float d1 = nv1 - 2.0f * acc1[r];
            int c0 = cc + mrow;
            int c1 = cc + 16 + mrow;
            if (d0 < minv || (d0 == minv && c0 < mini)) { minv = d0; mini = c0; }
            if (d1 < minv || (d1 == minv && c1 < mini)) { minv = d1; mini = c1; }
        }

        // -- publish staged buffer: drain own async ops, then block barrier --
        wait_async0();
        __syncthreads();
        buf ^= 1;
    }

    // ---- merge lane L with L+16 (same token, other row set) ----
    {
        float ov = __shfl_xor(minv, 16, 32);
        int   oi = __shfl_xor(mini, 16, 32);
        if (ov < minv || (ov == minv && oi < mini)) { minv = ov; mini = oi; }
    }
    if (lane < 16) IdxS[myTok + lane] = mini;
    __syncthreads();

    // ---- gather winning rows (f32 from original table): 2 threads/row ----
    {
        int row  = tid >> 1;
        int half = tid & 1;
        int code = IdxS[row];
        const float4* src = (const float4*)(E + (size_t)code * EMBED_DIM + half * 256);
        float4* dst = (float4*)(Out + (size_t)(tokBase + row) * EMBED_DIM + half * 256);
        #pragma unroll 8
        for (int i = 0; i < 64; ++i) dst[i] = src[i];
    }
}

// -------------------- host launcher --------------------
extern "C" void kernel_launch(void* const* d_in, const int* in_sizes, int n_in,
                              void* d_out, int out_size, void* d_ws, size_t ws_size,
                              hipStream_t stream) {
    const float* xs = (const float*)d_in[0];        // [32768, 512] f32
    const float* ew = (const float*)d_in[1];        // [8192, 512]  f32
    float* out      = (float*)d_out;                // [32768, 512] f32

    bf16_t* e16  = (bf16_t*)d_ws;                                   // 8 MB
    float* norms = (float*)((char*)d_ws + (size_t)N_EMBED * EMBED_DIM * sizeof(bf16_t));

    vq_prep<<<N_EMBED / 8, 256, 0, stream>>>(ew, e16, norms);

    size_t phase1 = (size_t)BLOCK_TOKENS * XSTRIDE * sizeof(bf16_t);            // 133,120
    size_t phase2 = 2 * (size_t)EBUF * sizeof(bf16_t) + BLOCK_TOKENS * sizeof(int);
    size_t smem = phase1 > phase2 ? phase1 : phase2;

    vq_main<<<N_TOKENS / BLOCK_TOKENS, 256, smem, stream>>>(xs, ew, e16, norms, out);
}
